// MHA_51092930953757
// MI455X (gfx1250) — matrix operand order repaired and checked by reading, not compiled, not run on hardware
//
#include <hip/hip_runtime.h>
#include <hip/hip_bf16.h>

// ---------------------------------------------------------------------------
// MHA for MI455X (gfx1250, wave32): bf16 WMMA pipeline + TDM-staged attention.
//   x:[2,2048,2048] f32, Wq/Wk/Wv/Wo:[2048,2048] f32  ->  out:[2,2048,2048] f32
// Workspace (bf16): Q[b,h,s,hd], K[b,h,s,hd], Vt[b,h,hd,s], O[b,s,h*hd]
//   = 4 * 8,388,608 * 2B = 64 MiB
// ---------------------------------------------------------------------------

#define HIDDEN   2048
#define HEADS    16
#define HEAD_DIM 128
#define SEQ      2048
#define BATCH    2
#define MROWS    (BATCH * SEQ)   // 4096

typedef __attribute__((ext_vector_type(16))) __bf16 bf16x16;
typedef __attribute__((ext_vector_type(8)))  __bf16 bf16x8;
typedef __attribute__((ext_vector_type(8)))  float  f32x8;
typedef __attribute__((ext_vector_type(4)))  unsigned int u32x4;
typedef __attribute__((ext_vector_type(8)))  int i32x8;
typedef __attribute__((ext_vector_type(4)))  int i32x4;

// f32 -> bf16 (round-to-nearest-even) as raw bits
static __device__ __forceinline__ unsigned short f2bfu(float f) {
    union { float f; unsigned u; } c; c.f = f;
    unsigned u = c.u;
    unsigned r = u + 0x7fffu + ((u >> 16) & 1u);
    return (unsigned short)(r >> 16);
}

static __device__ __forceinline__ bf16x8 ld8(const unsigned short* p) {
    union { uint4 q; bf16x8 v; } u;
    u.q = *reinterpret_cast<const uint4*>(p);
    return u.v;
}

static __device__ __forceinline__ bf16x16 ld16x(const unsigned short* p) {
    bf16x8 lo = ld8(p);
    bf16x8 hi = ld8(p + 8);
    return __builtin_shufflevector(lo, hi, 0,1,2,3,4,5,6,7,8,9,10,11,12,13,14,15);
}

static __device__ __forceinline__ bf16x16 combine(bf16x8 lo, bf16x8 hi) {
    return __builtin_shufflevector(lo, hi, 0,1,2,3,4,5,6,7,8,9,10,11,12,13,14,15);
}

static __device__ __forceinline__ f32x8 wmma_bf16(bf16x16 a, bf16x16 b, f32x8 c) {
    // (neg_a, A, neg_b, B, c_mod, C, reuse_a, reuse_b)
    return __builtin_amdgcn_wmma_f32_16x16x32_bf16(false, a, false, b, (short)0, c,
                                                   false, false);
}

// ---------------------------------------------------------------------------
// Tensor Data Mover: async 2D bf16 tile load Global -> LDS (D# per ISA §8).
// data_size=1 (2 bytes). Padding inserted every (pad_interval) DWORDs so LDS
// rows are 16B-aligned and bank-staggered.
// Toolchain here uses the 6-arg builtin: (g0, g1, g2, g3, g_extra, cpol).
// ---------------------------------------------------------------------------
static __device__ __forceinline__ void tdm_load_2d_bf16(
    unsigned int lds_off, const void* gptr,
    unsigned int tensor_d0, unsigned int tensor_d1, unsigned int d0_stride,
    unsigned int tile_d0, unsigned int tile_d1,
    unsigned int pad_interval_code, unsigned int pad_amount_code)
{
    unsigned long long ga = (unsigned long long)(uintptr_t)gptr;
    u32x4 g0;
    g0[0] = 1u;                                            // count=1, user descriptor
    g0[1] = lds_off;                                       // lds_addr (bytes)
    g0[2] = (unsigned int)(ga & 0xffffffffu);              // global_addr[31:0]
    g0[3] = (unsigned int)((ga >> 32) & 0x01ffffffu)       // global_addr[56:32]
          | 0x80000000u;                                   // type=2 ("image")
    i32x8 g1;
    g1[0] = (int)((1u << 16)                               // data_size = 2 bytes
                | (1u << 20)                               // pad_enable
                | (pad_interval_code << 22)
                | (pad_amount_code << 25));
    g1[1] = (int)((tensor_d0 & 0xffffu) << 16);            // tensor_dim0[15:0]
    g1[2] = (int)(((tensor_d0 >> 16) & 0xffffu) | ((tensor_d1 & 0xffffu) << 16));
    g1[3] = (int)(((tensor_d1 >> 16) & 0xffffu) | ((tile_d0 & 0xffffu) << 16));
    g1[4] = (int)(tile_d1 & 0xffffu);                      // tile_dim1, tile_dim2=0
    g1[5] = (int)d0_stride;                                // tensor_dim0_stride[31:0]
    g1[6] = 0;                                             // stride[47:32]=0, dim1_stride lo=0
    g1[7] = 0;
    i32x4 z4 = {0, 0, 0, 0};
    i32x8 z8 = {0, 0, 0, 0, 0, 0, 0, 0};
    __builtin_amdgcn_tensor_load_to_lds(g0, g1, z4, z4, z8, 0);
}

// ---------------------------------------------------------------------------
// GEMM: Y[m,n] = sum_k A[m,k] * W[n,k]   (A:[4096,2048], W:[2048,2048])
// Block = 256 threads = 8 waves, 128x128 tile; K staged 32 at a time into bf16
// LDS. Each wave owns a 32(M) x 64(N) sub-tile = 2x4 WMMA tiles.
// MODE 0: A=f32,  out -> bf16 [b,h,s,hd]      (Q / K projections)
// MODE 2: A=f32,  out -> bf16 [b,h,hd,s]      (V projection, transposed)
// MODE 3: A=bf16, out -> f32  [m,n]           (output projection)
// ---------------------------------------------------------------------------
#define BK   32
#define LDA  40   // 32 + 8 bf16 pad (80B row stride, 16B aligned)

template <int MODE>
__global__ __launch_bounds__(256)
void gemm128(const void* __restrict__ Aany, const float* __restrict__ W,
             void* __restrict__ Out)
{
    __shared__ unsigned short As[128 * LDA];
    __shared__ unsigned short Bs[128 * LDA];

    const int tid   = threadIdx.x;
    const int wave  = tid >> 5;
    const int lane  = tid & 31;
    const int half  = lane >> 4;
    const int l15   = lane & 15;
    const int waveM = wave & 3;
    const int waveN = wave >> 2;
    const int bm = blockIdx.y * 128;
    const int bn = blockIdx.x * 128;

    f32x8 acc[2][4];
#pragma unroll
    for (int mt = 0; mt < 2; ++mt)
#pragma unroll
        for (int nt = 0; nt < 4; ++nt)
            acc[mt][nt] = (f32x8){0.f,0.f,0.f,0.f,0.f,0.f,0.f,0.f};

    const int ldrow = tid >> 1;
    const int kseg  = (tid & 1) * 16;

    for (int kb = 0; kb < HIDDEN; kb += BK) {
        if constexpr (MODE == 3) {
            const unsigned short* A = (const unsigned short*)Aany;
            const unsigned short* src = A + (size_t)(bm + ldrow) * HIDDEN + kb + kseg;
            if (kb + BK < HIDDEN) __builtin_prefetch(src + BK, 0, 1);
            *reinterpret_cast<uint4*>(&As[ldrow * LDA + kseg])     = *reinterpret_cast<const uint4*>(src);
            *reinterpret_cast<uint4*>(&As[ldrow * LDA + kseg + 8]) = *reinterpret_cast<const uint4*>(src + 8);
        } else {
            const float* A = (const float*)Aany;
            const float* src = A + (size_t)(bm + ldrow) * HIDDEN + kb + kseg;
            if (kb + BK < HIDDEN) __builtin_prefetch(src + BK, 0, 1);
            float4 f0 = *reinterpret_cast<const float4*>(src);
            float4 f1 = *reinterpret_cast<const float4*>(src + 4);
            float4 f2 = *reinterpret_cast<const float4*>(src + 8);
            float4 f3 = *reinterpret_cast<const float4*>(src + 12);
            union { unsigned short s[8]; uint4 q; } p0, p1;
            p0.s[0]=f2bfu(f0.x); p0.s[1]=f2bfu(f0.y); p0.s[2]=f2bfu(f0.z); p0.s[3]=f2bfu(f0.w);
            p0.s[4]=f2bfu(f1.x); p0.s[5]=f2bfu(f1.y); p0.s[6]=f2bfu(f1.z); p0.s[7]=f2bfu(f1.w);
            p1.s[0]=f2bfu(f2.x); p1.s[1]=f2bfu(f2.y); p1.s[2]=f2bfu(f2.z); p1.s[3]=f2bfu(f2.w);
            p1.s[4]=f2bfu(f3.x); p1.s[5]=f2bfu(f3.y); p1.s[6]=f2bfu(f3.z); p1.s[7]=f2bfu(f3.w);
            *reinterpret_cast<uint4*>(&As[ldrow * LDA + kseg])     = p0.q;
            *reinterpret_cast<uint4*>(&As[ldrow * LDA + kseg + 8]) = p1.q;
        }
        {
            const float* src = W + (size_t)(bn + ldrow) * HIDDEN + kb + kseg;
            if (kb + BK < HIDDEN) __builtin_prefetch(src + BK, 0, 1);
            float4 f0 = *reinterpret_cast<const float4*>(src);
            float4 f1 = *reinterpret_cast<const float4*>(src + 4);
            float4 f2 = *reinterpret_cast<const float4*>(src + 8);
            float4 f3 = *reinterpret_cast<const float4*>(src + 12);
            union { unsigned short s[8]; uint4 q; } p0, p1;
            p0.s[0]=f2bfu(f0.x); p0.s[1]=f2bfu(f0.y); p0.s[2]=f2bfu(f0.z); p0.s[3]=f2bfu(f0.w);
            p0.s[4]=f2bfu(f1.x); p0.s[5]=f2bfu(f1.y); p0.s[6]=f2bfu(f1.z); p0.s[7]=f2bfu(f1.w);
            p1.s[0]=f2bfu(f2.x); p1.s[1]=f2bfu(f2.y); p1.s[2]=f2bfu(f2.z); p1.s[3]=f2bfu(f2.w);
            p1.s[4]=f2bfu(f3.x); p1.s[5]=f2bfu(f3.y); p1.s[6]=f2bfu(f3.z); p1.s[7]=f2bfu(f3.w);
            *reinterpret_cast<uint4*>(&Bs[ldrow * LDA + kseg])     = p0.q;
            *reinterpret_cast<uint4*>(&Bs[ldrow * LDA + kseg + 8]) = p1.q;
        }
        __syncthreads();

        bf16x16 afr[2], bfr[4];
#pragma unroll
        for (int mt = 0; mt < 2; ++mt) {
            const int row  = waveM * 32 + mt * 16 + l15;
            const int base = half * 16;
            bf16x8 lo = *reinterpret_cast<const bf16x8*>(&As[row * LDA + base]);
            bf16x8 hi = *reinterpret_cast<const bf16x8*>(&As[row * LDA + base + 8]);
            afr[mt] = combine(lo, hi);
        }
#pragma unroll
        for (int nt = 0; nt < 4; ++nt) {
            const int row  = waveN * 64 + nt * 16 + l15;
            const int base = half * 16;
            bf16x8 lo = *reinterpret_cast<const bf16x8*>(&Bs[row * LDA + base]);
            bf16x8 hi = *reinterpret_cast<const bf16x8*>(&Bs[row * LDA + base + 8]);
            bfr[nt] = combine(lo, hi);
        }
#pragma unroll
        for (int mt = 0; mt < 2; ++mt)
#pragma unroll
            for (int nt = 0; nt < 4; ++nt)
                acc[mt][nt] = wmma_bf16(afr[mt], bfr[nt], acc[mt][nt]);
        __syncthreads();
    }

#pragma unroll
    for (int mt = 0; mt < 2; ++mt) {
#pragma unroll
        for (int nt = 0; nt < 4; ++nt) {
#pragma unroll
            for (int r = 0; r < 8; ++r) {
                const int mg = bm + waveM * 32 + mt * 16 + r + half * 8;
                const int ng = bn + waveN * 64 + nt * 16 + l15;
                const float val = acc[mt][nt][r];
                if constexpr (MODE == 3) {
                    ((float*)Out)[(size_t)mg * HIDDEN + ng] = val;
                } else {
                    const int b = mg >> 11, srow = mg & 2047;
                    const int hh = ng >> 7, col = ng & 127;
                    size_t idx;
                    if constexpr (MODE == 2)
                        idx = ((size_t)(b * HEADS + hh) * HEAD_DIM + col) * SEQ + srow;
                    else
                        idx = ((size_t)(b * HEADS + hh) * SEQ + srow) * HEAD_DIM + col;
                    ((unsigned short*)Out)[idx] = f2bfu(val);
                }
            }
        }
    }
}

// ---------------------------------------------------------------------------
// Flash attention with TDM double-buffered K/V staging.
// Block = 8 waves x 16 q rows = 128 q rows; grid = (SEQ/128, BATCH*HEADS).
// K tile: 32 keys x 128 hd bf16 -> LDS rows of 272B (64DW data + 4DW pad)
// V tile: 128 hd x 32 keys bf16 -> LDS rows of  80B (16DW data + 4DW pad)
// Wave 0 drives the TDM; TENSORcnt overlaps tile t+1 DMA with tile t compute.
// ---------------------------------------------------------------------------
#define KT_LD 136   // ushorts per K-tile LDS row (272B)
#define VT_LD 40    // ushorts per V-tile LDS row (80B)

__global__ __launch_bounds__(256)
void attn_kernel(const unsigned short* __restrict__ Q,
                 const unsigned short* __restrict__ K,
                 const unsigned short* __restrict__ Vt,
                 unsigned short* __restrict__ O)
{
    __shared__ unsigned short KT[2][32 * KT_LD];
    __shared__ unsigned short VT[2][128 * VT_LD];
    __shared__ unsigned short Pt[8][16][40];     // per-wave 16x32 P tile (+pad)

    const int tid  = threadIdx.x;
    const int wave = tid >> 5;
    const int lane = tid & 31;
    const int half = lane >> 4;
    const int l15  = lane & 15;
    const int bh   = blockIdx.y;
    const int q0   = blockIdx.x * 128 + wave * 16;

    const unsigned short* Qb = Q  + (size_t)bh * SEQ * HEAD_DIM;
    const unsigned short* Kb = K  + (size_t)bh * SEQ * HEAD_DIM;
    const unsigned short* Vb = Vt + (size_t)bh * HEAD_DIM * SEQ;

    // Escape the TDM-written LDS buffers so the optimizer cannot treat them as
    // never-stored (the TDM builtin carries no pointer argument).
    asm volatile("" :: "s"((uintptr_t)&KT[0][0]), "s"((uintptr_t)&VT[0][0]) : "memory");
    if ((int)blockDim.y == 12345) { KT[0][0] = 1; VT[0][0] = 1; }  // never taken

    // Q row-block in registers: 4 A-fragments of 16x32 over hd
    bf16x16 qa[4];
#pragma unroll
    for (int c = 0; c < 4; ++c) {
        const unsigned short* p = Qb + (size_t)(q0 + l15) * HEAD_DIM + c * 32 + half * 16;
        qa[c] = combine(ld8(p), ld8(p + 8));
    }

    float m[8], l[8];
    f32x8 o[8];
#pragma unroll
    for (int r = 0; r < 8; ++r) { m[r] = -__builtin_inff(); l[r] = 0.f; }
#pragma unroll
    for (int nt = 0; nt < 8; ++nt) o[nt] = (f32x8){0.f,0.f,0.f,0.f,0.f,0.f,0.f,0.f};

    const float sm = 0.08838834764831843f;   // 128^-0.5

    // Prologue: wave0 kicks off tile 0 (K tile + V tile = 2 TDM ops).
    if (wave == 0) {
        tdm_load_2d_bf16((unsigned int)(uintptr_t)&KT[0][0], Kb,
                         HEAD_DIM, SEQ, HEAD_DIM, HEAD_DIM, 32, /*pad*/5, 3);
        tdm_load_2d_bf16((unsigned int)(uintptr_t)&VT[0][0], Vb,
                         SEQ, HEAD_DIM, SEQ, 32, HEAD_DIM, /*pad*/3, 3);
    }

    for (int kb0 = 0; kb0 < SEQ; kb0 += 32) {
        const int buf = (kb0 >> 5) & 1;
        __syncthreads();   // all waves done reading both buffers (previous iter)
        if (wave == 0) {
            if (kb0 + 32 < SEQ) {
                const int nb = buf ^ 1, kn = kb0 + 32;
                tdm_load_2d_bf16((unsigned int)(uintptr_t)&KT[nb][0],
                                 Kb + (size_t)kn * HEAD_DIM,
                                 HEAD_DIM, SEQ, HEAD_DIM, HEAD_DIM, 32, 5, 3);
                tdm_load_2d_bf16((unsigned int)(uintptr_t)&VT[nb][0],
                                 Vb + kn,
                                 SEQ, HEAD_DIM, SEQ, 32, HEAD_DIM, 3, 3);
                __builtin_amdgcn_s_wait_tensorcnt(2);  // tile t done, t+1 in flight
            } else {
                __builtin_amdgcn_s_wait_tensorcnt(0);
            }
        }
        __syncthreads();   // publish buffer `buf`
        const unsigned short* kt = &KT[buf][0];
        const unsigned short* vt = &VT[buf][0];

        // ---- S = Q @ K^T for this 16x32 key block (two N tiles) ----
        f32x8 s0 = (f32x8){0.f,0.f,0.f,0.f,0.f,0.f,0.f,0.f};
        f32x8 s1 = s0;
#pragma unroll
        for (int c = 0; c < 4; ++c) {
            const unsigned short* pk0 = kt + (l15)      * KT_LD + c * 32 + half * 16;
            const unsigned short* pk1 = kt + (16 + l15) * KT_LD + c * 32 + half * 16;
            s0 = wmma_bf16(qa[c], ld16x(pk0), s0);
            s1 = wmma_bf16(qa[c], ld16x(pk1), s1);
        }

        // ---- online softmax (row = r + 8*half, cols across 16-lane group) ----
        float rm[8];
#pragma unroll
        for (int r = 0; r < 8; ++r) {
            s0[r] *= sm; s1[r] *= sm;
            float t = fmaxf(s0[r], s1[r]);
            t = fmaxf(t, __shfl_xor(t, 1, 16));
            t = fmaxf(t, __shfl_xor(t, 2, 16));
            t = fmaxf(t, __shfl_xor(t, 4, 16));
            t = fmaxf(t, __shfl_xor(t, 8, 16));
            rm[r] = t;
        }
#pragma unroll
        for (int r = 0; r < 8; ++r) {
            const float mn = fmaxf(m[r], rm[r]);
            const float sc = __expf(m[r] - mn);
            m[r] = mn;
            l[r] *= sc;
#pragma unroll
            for (int nt = 0; nt < 8; ++nt) o[nt][r] *= sc;
            const float p0 = __expf(s0[r] - mn);
            const float p1 = __expf(s1[r] - mn);
            float rs = p0 + p1;
            rs += __shfl_xor(rs, 1, 16);
            rs += __shfl_xor(rs, 2, 16);
            rs += __shfl_xor(rs, 4, 16);
            rs += __shfl_xor(rs, 8, 16);
            l[r] += rs;
            // C-layout -> LDS (row-major 16x32 bf16) for A-layout reload
            Pt[wave][r + half * 8][l15]      = f2bfu(p0);
            Pt[wave][r + half * 8][16 + l15] = f2bfu(p1);
        }
        // wave-private LDS region: DS ops are in-order per wave; wait for stores
        asm volatile("s_wait_dscnt 0" ::: "memory");

        bf16x16 pa;
        {
            bf16x8 lo = *reinterpret_cast<const bf16x8*>(&Pt[wave][l15][half * 16]);
            bf16x8 hi = *reinterpret_cast<const bf16x8*>(&Pt[wave][l15][half * 16 + 8]);
            pa = combine(lo, hi);
        }

        // ---- O += P @ V  (V^T tile in LDS: contiguous keys per hd row) ----
#pragma unroll
        for (int nt = 0; nt < 8; ++nt) {
            const unsigned short* pv = vt + (nt * 16 + l15) * VT_LD + half * 16;
            o[nt] = wmma_bf16(pa, ld16x(pv), o[nt]);
        }
    }

    // ---- normalize + write O[b, s, h*hd] bf16 ----
    const int b = bh >> 4, h = bh & 15;
#pragma unroll
    for (int r = 0; r < 8; ++r) {
        const float inv = 1.0f / l[r];
        const int q = q0 + r + half * 8;
#pragma unroll
        for (int nt = 0; nt < 8; ++nt) {
            O[(size_t)(b * SEQ + q) * HIDDEN + h * HEAD_DIM + nt * 16 + l15] =
                f2bfu(o[nt][r] * inv);
        }
    }
}

// ---------------------------------------------------------------------------
extern "C" void kernel_launch(void* const* d_in, const int* in_sizes, int n_in,
                              void* d_out, int out_size, void* d_ws, size_t ws_size,
                              hipStream_t stream) {
    (void)in_sizes; (void)n_in; (void)out_size; (void)ws_size;
    const float* x  = (const float*)d_in[0];
    const float* Wq = (const float*)d_in[1];
    const float* Wk = (const float*)d_in[2];
    const float* Wv = (const float*)d_in[3];
    const float* Wo = (const float*)d_in[4];

    const size_t QSZ = (size_t)BATCH * HEADS * SEQ * HEAD_DIM;   // 8,388,608 elems
    unsigned short* Qw = (unsigned short*)d_ws;
    unsigned short* Kw = Qw + QSZ;
    unsigned short* Vt = Kw + QSZ;
    unsigned short* Ow = Vt + QSZ;

    dim3 ggrid(HIDDEN / 128, MROWS / 128);   // (16, 32)
    gemm128<0><<<ggrid, 256, 0, stream>>>(x, Wq, Qw);
    gemm128<0><<<ggrid, 256, 0, stream>>>(x, Wk, Kw);
    gemm128<2><<<ggrid, 256, 0, stream>>>(x, Wv, Vt);

    dim3 agrid(SEQ / 128, BATCH * HEADS);    // (16, 32)
    attn_kernel<<<agrid, 256, 0, stream>>>(Qw, Kw, Vt, Ow);

    gemm128<3><<<ggrid, 256, 0, stream>>>(Ow, Wo, (float*)d_out);
}
